// EdgePredictor_69260642615902
// MI455X (gfx1250) — matrix operand order repaired and checked by reading, not compiled, not run on hardware
//
#include <hip/hip_runtime.h>
#include <hip/hip_bf16.h>
#include <math.h>

// ---------------- problem constants ----------------
#define N_NODES   2048
#define NODE_DIM  128
#define HID       128
#define HID2      64
#define NPAIRS    2096128            // 2048*2047/2
#define NTILES    (NPAIRS / 16)      // 131008, exact

typedef __attribute__((ext_vector_type(16))) _Float16 v16h;
typedef __attribute__((ext_vector_type(8)))  _Float16 v8h;
typedef __attribute__((ext_vector_type(4)))  _Float16 v4h;
typedef __attribute__((ext_vector_type(8)))  float    v8f;

// prefix count of upper-triangular (k=1) pairs before row i: C(i) = i*(2N-1-i)/2
__device__ __forceinline__ long long triC(long long i) { return i * (4095LL - i) / 2; }

// ---------------- kernel 1: h = relu(X @ Wt + bt) ----------------
__global__ void node_transform_kernel(const float* __restrict__ X,
                                      const float* __restrict__ Wt,
                                      const float* __restrict__ bt,
                                      float* __restrict__ h) {
    __shared__ float xrow[NODE_DIM];
    const int row = blockIdx.x;
    const int c   = threadIdx.x;                 // 128 threads
    xrow[c] = X[row * NODE_DIM + c];
    __syncthreads();
    float acc = bt[c];
#pragma unroll 8
    for (int k = 0; k < NODE_DIM; ++k)
        acc = fmaf(xrow[k], Wt[k * HID + c], acc);   // coalesced over c
    h[row * HID + c] = fmaxf(acc, 0.f);
}

// ---------------- kernel 2: A = h@W1[:128] + b1 ;  B = h@W1[128:] ----------------
__global__ void pair_precompute_kernel(const float* __restrict__ h,
                                       const float* __restrict__ W1,
                                       const float* __restrict__ b1,
                                       float* __restrict__ Ap,
                                       float* __restrict__ Bp) {
    __shared__ float hrow[HID];
    const int row = blockIdx.x;
    const int c   = threadIdx.x;                 // 128 threads
    hrow[c] = h[row * HID + c];
    __syncthreads();
    float a = b1[c], b = 0.f;
#pragma unroll 8
    for (int k = 0; k < HID; ++k) {
        const float hv = hrow[k];
        a = fmaf(hv, W1[k * HID + c], a);
        b = fmaf(hv, W1[(HID + k) * HID + c], b);
    }
    Ap[row * HID + c] = a;
    Bp[row * HID + c] = b;
}

// ---------------- kernel 3: per-pair LN + WMMA MLP + scatter ----------------
// One wave (32 lanes) processes a tile of 16 pairs. Lanes L and L+16 co-own
// pair-row (L&15): each handles 64 of the 128 z-columns. W2 B-fragments are
// hoisted into registers (loop-invariant) so the WMMA stage issues with only
// 2 A-fragment DS loads per K-step.
__global__ __launch_bounds__(256) void edge_pair_kernel(
    const float* __restrict__ Ap, const float* __restrict__ Bp,
    const float* __restrict__ lnG, const float* __restrict__ lnB,
    const float* __restrict__ W2, const float* __restrict__ b2,
    const float* __restrict__ W3, const float* __restrict__ b3,
    float* __restrict__ probs, float* __restrict__ pidx, float* __restrict__ adj)
{
    // static LDS: 16KB W2-pack + 32KB z-stage + 8KB partials + 1KB LN params
    __shared__ __align__(32) _Float16 w2t[HID2 * HID];   // [n][k], k-contiguous
    __shared__ __align__(32) _Float16 zsh[8][16 * HID];  // per-wave normalized z (f16)
    __shared__ __align__(16) float    part[8][8 * 32];   // per-wave W3 partials
    __shared__ __align__(16) float    lgs[HID], lbs[HID];

    const int tid = threadIdx.x;

    // pack W2 transposed (f32 -> f16) so B fragments load as one contiguous v16h
    for (int e = tid; e < HID2 * HID; e += 256) {
        const int n = e >> 7, k = e & 127;
        w2t[e] = (_Float16)W2[k * HID2 + n];
    }
    if (tid < HID) { lgs[tid] = lnG[tid]; lbs[tid] = lnB[tid]; }
    __syncthreads();   // only block-wide sync; waves are independent afterwards

    const int  lane    = tid & 31;
    const int  l16     = lane & 15;
    const int  wv      = tid >> 5;
    const int  colbase = (lane >> 4) * 64;   // 0 or 64: which half of the 128 cols
    const bool hiHalf  = lane >= 16;

    float b2v[4], w3v[4];
#pragma unroll
    for (int nt = 0; nt < 4; ++nt) {
        b2v[nt] = b2[nt * 16 + l16];
        w3v[nt] = W3[nt * 16 + l16];
    }
    const float b3s = b3[0];

    _Float16* zw = &zsh[wv][0];
    float*    pw = &part[wv][0];

    // ---- hoist all 16 loop-invariant B fragments (packed W2) into registers ----
    // B fragment layout: lane<16 -> N = nt*16+l16, K = ks*32 + 0..15 ;
    //                    lane>=16 -> same N,      K = ks*32 + 16..31
    v16h bfrag[16];
#pragma unroll
    for (int ks = 0; ks < 4; ++ks) {
        const int kb = ks * 32 + (hiHalf ? 16 : 0);
#pragma unroll
        for (int nt = 0; nt < 4; ++nt)
            bfrag[ks * 4 + nt] = *(const v16h*)&w2t[(nt * 16 + l16) * HID + kb];
    }
    asm volatile("s_wait_dscnt 0" ::: "memory");

    const int wgid = blockIdx.x * (blockDim.x >> 5) + wv;
    const int nW   = gridDim.x * (blockDim.x >> 5);

    for (int t = wgid; t < NTILES; t += nW) {       // uniform per wave -> EXEC full
        // ---- invert triu(k=1) linear index -> (i, j) ----
        const long long p = (long long)t * 16 + l16;
        const double disc = 4095.0 * 4095.0 - 8.0 * (double)p;
        long long i = (long long)((4095.0 - sqrt(disc)) * 0.5);
        if (i < 0) i = 0;
        while (triC(i + 1) <= p) ++i;
        while (triC(i) > p)      --i;
        const long long j = p - triC(i) + i + 1;

        // ---- pass 1: z = A[i] + B[j] (layer-1 decomposition), LN stats ----
        const float4* arow = (const float4*)(Ap + (size_t)i * HID + colbase);
        const float4* brow = (const float4*)(Bp + (size_t)j * HID + colbase);
        float4 z[16];
        float s = 0.f, s2 = 0.f;
#pragma unroll
        for (int q = 0; q < 16; ++q) {
            const float4 av = arow[q], bv = brow[q];
            float4 zv;
            zv.x = av.x + bv.x; zv.y = av.y + bv.y;
            zv.z = av.z + bv.z; zv.w = av.w + bv.w;
            z[q] = zv;
            s  += zv.x + zv.y + zv.z + zv.w;
            s2  = fmaf(zv.x, zv.x, s2); s2 = fmaf(zv.y, zv.y, s2);
            s2  = fmaf(zv.z, zv.z, s2); s2 = fmaf(zv.w, zv.w, s2);
        }
        s  += __shfl_xor(s, 16);     // combine the two half-rows
        s2 += __shfl_xor(s2, 16);
        const float mean = s * (1.f / 128.f);
        const float var  = s2 * (1.f / 128.f) - mean * mean;
        const float rstd = rsqrtf(fmaxf(var, 0.f) + 1e-5f);

        // ---- normalize, relu, f16, stage row-major into per-wave LDS ----
        const float4* lg4 = (const float4*)(lgs + colbase);
        const float4* lb4 = (const float4*)(lbs + colbase);
#pragma unroll
        for (int q = 0; q < 16; ++q) {
            const float4 g = lg4[q], bb = lb4[q], zv = z[q];
            const float x0 = fmaxf(fmaf((zv.x - mean) * rstd, g.x, bb.x), 0.f);
            const float x1 = fmaxf(fmaf((zv.y - mean) * rstd, g.y, bb.y), 0.f);
            const float x2 = fmaxf(fmaf((zv.z - mean) * rstd, g.z, bb.z), 0.f);
            const float x3 = fmaxf(fmaf((zv.w - mean) * rstd, g.w, bb.w), 0.f);
            v4h hv;
            hv[0] = (_Float16)x0; hv[1] = (_Float16)x1;
            hv[2] = (_Float16)x2; hv[3] = (_Float16)x3;
            *(v4h*)&zw[l16 * HID + colbase + q * 4] = hv;
        }
        // per-wave DS ops are in-order; this is a compiler fence + hard wait
        asm volatile("s_wait_dscnt 0" ::: "memory");

        // ---- [16,128] @ [128,64] via v_wmma_f32_16x16x32_f16 ----
        v8f acc[4] = {};
#pragma unroll
        for (int ks = 0; ks < 4; ++ks) {
            const int k0 = ks * 32;
            // A fragment layout (matches B): lane<16: K k0..k0+15 ;
            // lane>=16: K k0+16..k0+31
            const int ao = hiHalf ? 16 : 0;
            const v8h alo = *(const v8h*)&zw[l16 * HID + k0 + ao];
            const v8h ahi = *(const v8h*)&zw[l16 * HID + k0 + 8 + ao];
            v16h a;
#pragma unroll
            for (int e = 0; e < 8; ++e) { a[e] = alo[e]; a[8 + e] = ahi[e]; }
#pragma unroll
            for (int nt = 0; nt < 4; ++nt) {
                acc[nt] = __builtin_amdgcn_wmma_f32_16x16x32_f16(
                    false, a, false, bfrag[ks * 4 + nt], (short)0, acc[nt],
                    false, false);
            }
        }

        // ---- epilogue: +b2, relu, partial dot with W3 ----
        // acc[nt] element v = y[row = v + (hiHalf?8:0)][col = nt*16 + l16]
#pragma unroll
        for (int v = 0; v < 8; ++v) {
            float d3 = 0.f;
#pragma unroll
            for (int nt = 0; nt < 4; ++nt) {
                const float y = fmaxf(acc[nt][v] + b2v[nt], 0.f);
                d3 = fmaf(y, w3v[nt], d3);
            }
            pw[v * 32 + lane] = d3;   // partials: row-half v, 16 lanes each half
        }
        asm volatile("s_wait_dscnt 0" ::: "memory");

        if (lane < 16) {              // lane r finalizes pair-row r
            const int rb = (lane < 8) ? 0 : 16;
            const float4* pr = (const float4*)&pw[(lane & 7) * 32 + rb];
            const float4 q0 = pr[0], q1 = pr[1], q2 = pr[2], q3 = pr[3];
            const float logit = b3s
                + q0.x + q0.y + q0.z + q0.w + q1.x + q1.y + q1.z + q1.w
                + q2.x + q2.y + q2.z + q2.w + q3.x + q3.y + q3.z + q3.w;
            const float prob = 1.f / (1.f + __expf(-logit));
            probs[p] = prob;
            // pair_index is int32 in the reference; emitted here as exact float values
            pidx[p]           = (float)i;
            pidx[NPAIRS + p]  = (float)j;
            adj[(size_t)i * N_NODES + j] = prob;
            adj[(size_t)j * N_NODES + i] = prob;
        }
    }
}

// ---------------- kernel 4: zero the soft_adj diagonal ----------------
__global__ void diag_zero_kernel(float* __restrict__ adj) {
    const int d = blockIdx.x * blockDim.x + threadIdx.x;
    if (d < N_NODES) adj[(size_t)d * N_NODES + d] = 0.f;
}

// ---------------- launcher ----------------
extern "C" void kernel_launch(void* const* d_in, const int* in_sizes, int n_in,
                              void* d_out, int out_size, void* d_ws, size_t ws_size,
                              hipStream_t stream) {
    (void)in_sizes; (void)n_in; (void)out_size; (void)ws_size;
    const float* X  = (const float*)d_in[0];
    const float* Wt = (const float*)d_in[1];
    const float* bt = (const float*)d_in[2];
    const float* W1 = (const float*)d_in[3];
    const float* b1 = (const float*)d_in[4];
    const float* lg = (const float*)d_in[5];
    const float* lb = (const float*)d_in[6];
    const float* W2 = (const float*)d_in[7];
    const float* b2 = (const float*)d_in[8];
    const float* W3 = (const float*)d_in[9];
    const float* b3 = (const float*)d_in[10];

    float* ws = (float*)d_ws;                        // 3 MB scratch used
    float* Ap = ws;                                  // [2048,128]
    float* Bp = ws + (size_t)N_NODES * HID;          // [2048,128]
    float* h  = ws + (size_t)2 * N_NODES * HID;      // [2048,128]

    float* out   = (float*)d_out;
    float* probs = out;                              // [P]
    float* pidx  = out + NPAIRS;                     // [2,P]
    float* adj   = out + (size_t)3 * NPAIRS;         // [2048,2048]

    node_transform_kernel<<<N_NODES, 128, 0, stream>>>(X, Wt, bt, h);
    pair_precompute_kernel<<<N_NODES, 128, 0, stream>>>(h, W1, b1, Ap, Bp);
    diag_zero_kernel<<<(N_NODES + 255) / 256, 256, 0, stream>>>(adj);
    edge_pair_kernel<<<1024, 256, 0, stream>>>(Ap, Bp, lg, lb, W2, b2, W3, b3,
                                               probs, pidx, adj);
}
